// TripletColbertLoss_52664888983644
// MI455X (gfx1250) — matrix operand order repaired and checked by reading, not compiled, not run on hardware
//
#include <hip/hip_runtime.h>
#include <hip/hip_bf16.h>

// CDNA5 / gfx1250: wave32, bf16 WMMA 16x16x32, f32 accumulate.
typedef __attribute__((ext_vector_type(16))) __bf16 v16bf;
typedef __attribute__((ext_vector_type(8)))  float  v8f;

#define B_DIM 256
#define Q_DIM 256
#define K_DIM 512
#define D_DIM 128
#define NT    (K_DIM / 16)       // 32 doc-token tiles
#define ROWB  272                // 256B of bf16 + 16B pad -> bank-tiled b128 reads
#define MARGIN_F 0.2f

// bf16 WMMA operand as raw 32B (8 packed words / 2 uint4).
union BV { v16bf v; unsigned u[8]; uint4 q[2]; };

// Split two f32 into packed bf16 hi-word + lo-word (truncation + residual).
// q·d recovered as hi·hi + hi·lo + lo·hi, ~2^-15 relative error.
__device__ __forceinline__ void split_pack2(float f0, float f1,
                                            unsigned &hi, unsigned &lo) {
  unsigned u0 = __float_as_uint(f0);
  unsigned u1 = __float_as_uint(f1);
  unsigned h0 = u0 & 0xFFFF0000u;
  unsigned h1 = u1 & 0xFFFF0000u;
  hi = (u0 >> 16) | h1;
  float r0 = f0 - __uint_as_float(h0);
  float r1 = f1 - __uint_as_float(h1);
  lo = (__float_as_uint(r0) >> 16) | (__float_as_uint(r1) & 0xFFFF0000u);
}

__device__ __forceinline__ void pack8(const float4 &x0, const float4 &x1,
                                      BV &hi, BV &lo, int w0) {
  split_pack2(x0.x, x0.y, hi.u[w0 + 0], lo.u[w0 + 0]);
  split_pack2(x0.z, x0.w, hi.u[w0 + 1], lo.u[w0 + 1]);
  split_pack2(x1.x, x1.y, hi.u[w0 + 2], lo.u[w0 + 2]);
  split_pack2(x1.z, x1.w, hi.u[w0 + 3], lo.u[w0 + 3]);
}

__global__ void __launch_bounds__(256)
colbert_init_kernel(float* acc) {
  if (threadIdx.x < 2) acc[threadIdx.x] = 0.0f;
}

// Block = 8 waves: 4 q-tiles x {p,n}. All waves of a matrix share LDS-staged,
// pre-split doc tiles (conversion done once per block, not once per wave).
__global__ void __launch_bounds__(256)
colbert_maxsim_kernel(const float* __restrict__ q,
                      const float* __restrict__ p,
                      const float* __restrict__ n,
                      float* __restrict__ acc) {
  // [buf][mat][hi=0/lo=1][token][ROWB bytes] = 34,816 B
  __shared__ __align__(16) unsigned char lds[2][2][2][16][ROWB];

  const int tid   = threadIdx.x;
  const int lane  = tid & 31;
  const int wave  = tid >> 5;
  const int qtile = blockIdx.x * 4 + (wave >> 1);
  const int mat   = wave & 1;                 // 0 = pos, 1 = neg
  const int b     = blockIdx.x >> 2;          // 4 blocks per batch (uniform)
  const int qrow0 = (qtile & 15) << 4;

  // ---- producer mapping: threads 0-127 stage p, 128-255 stage n ----
  const int smat = tid >> 7;
  const int sidx = tid & 127;
  const int stok = sidx & 15;                 // doc token within tile
  const int sfg  = sidx >> 4;                 // 16-feature group (0..7)
  const float* __restrict__ sdoc = smat ? n : p;
  const float* __restrict__ sbase =
      sdoc + (size_t)b * K_DIM * D_DIM + (size_t)stok * D_DIM + sfg * 16;

  auto stage = [&](int kt, int buf) {
    const float4* __restrict__ s4 =
        (const float4*)(sbase + (size_t)kt * 16 * D_DIM);  // 64B aligned
    float4 x0 = s4[0], x1 = s4[1], x2 = s4[2], x3 = s4[3];
    BV h, l;
    pack8(x0, x1, h, l, 0);
    pack8(x2, x3, h, l, 4);
    uint4* hd = (uint4*)&lds[buf][smat][0][stok][sfg * 32];
    uint4* ld = (uint4*)&lds[buf][smat][1][stok][sfg * 32];
    hd[0] = h.q[0]; hd[1] = h.q[1];
    ld[0] = l.q[0]; ld[1] = l.q[1];
  };

  // ---- resident A tile: q[b, qrow0:+16, 0:128], split to bf16 hi/lo ----
  // A k order used here (contiguous, same as B): lane<16 row=lane K=c*32+0..15;
  // lane>=16 row=lane-16 K=c*32+16..31.
  const int arow = lane & 15;
  const float4* __restrict__ q4 =
      (const float4*)(q + ((size_t)b * Q_DIM + qrow0 + arow) * D_DIM);
  const int kofs = (lane < 16) ? 0 : 16;
  v16bf a_hi[4], a_lo[4];
#pragma unroll
  for (int c = 0; c < 4; ++c) {
    const int base = c * 32 + kofs;
    float4 x0 = q4[(base >> 2) + 0];
    float4 x1 = q4[(base >> 2) + 1];
    float4 y0 = q4[((base + 8) >> 2) + 0];
    float4 y1 = q4[((base + 8) >> 2) + 1];
    BV ah, al;
    pack8(x0, x1, ah, al, 0);
    pack8(y0, y1, ah, al, 4);
    a_hi[c] = ah.v;
    a_lo[c] = al.v;
  }

  float m[8];
#pragma unroll
  for (int r = 0; r < 8; ++r) m[r] = -3.402823466e38f;

  // B k order used here (contiguous, same as A): lane<16 -> col n=lane, K=0..15; lane>=16 ->
  // col n=lane-16, K=16..31. col = doc token, K = feature.
  const int col  = lane & 15;
  const int fofs = (lane < 16) ? 0 : 16;

  stage(0, 0);
  __syncthreads();

  for (int kt = 0; kt < NT; ++kt) {
    const int buf = kt & 1;
    if (kt + 1 < NT) stage(kt + 1, buf ^ 1);  // pipeline next tile

    v8f c1 = {};  // hi*hi chain
    v8f c2 = {};  // hi*lo + lo*hi correction chain
#pragma unroll
    for (int c = 0; c < 4; ++c) {
      const uint4* __restrict__ hp =
          (const uint4*)&lds[buf][mat][0][col][c * 64 + fofs * 2];
      const uint4* __restrict__ lp =
          (const uint4*)&lds[buf][mat][1][col][c * 64 + fofs * 2];
      BV bh, bl;
      bh.q[0] = hp[0]; bh.q[1] = hp[1];
      bl.q[0] = lp[0]; bl.q[1] = lp[1];
      c1 = __builtin_amdgcn_wmma_f32_16x16x32_bf16(false, a_hi[c], false, bh.v,
                                                   (short)0, c1, false, false);
      c2 = __builtin_amdgcn_wmma_f32_16x16x32_bf16(false, a_hi[c], false, bl.v,
                                                   (short)0, c2, false, false);
      c2 = __builtin_amdgcn_wmma_f32_16x16x32_bf16(false, a_lo[c], false, bh.v,
                                                   (short)0, c2, false, false);
    }
    // C/D layout: VGPR r, lanes 0-15 -> (M=r, N=lane); lanes 16-31 -> M=8+r.
#pragma unroll
    for (int r = 0; r < 8; ++r) m[r] = fmaxf(m[r], c1[r] + c2[r]);

    __syncthreads();  // staged writes visible; reads done before overwrite
  }

  // Row max across the 16 lanes of each half (xor 1,2,4,8 stays in-half).
  float s = 0.0f;
#pragma unroll
  for (int r = 0; r < 8; ++r) {
    float v = m[r];
    v = fmaxf(v, __shfl_xor(v, 1, 32));
    v = fmaxf(v, __shfl_xor(v, 2, 32));
    v = fmaxf(v, __shfl_xor(v, 4, 32));
    v = fmaxf(v, __shfl_xor(v, 8, 32));
    s += v;
  }
  if (lane == 0 || lane == 16) atomicAdd(&acc[mat], s);
}

__global__ void colbert_finalize_kernel(const float* __restrict__ acc,
                                        float* __restrict__ out) {
  float l = MARGIN_F + acc[1] - acc[0];
  out[0] = l > 0.0f ? l : 0.0f;
}

extern "C" void kernel_launch(void* const* d_in, const int* in_sizes, int n_in,
                              void* d_out, int out_size, void* d_ws,
                              size_t ws_size, hipStream_t stream) {
  const float* q = (const float*)d_in[0];
  const float* p = (const float*)d_in[1];
  const float* n = (const float*)d_in[2];
  float* acc = (float*)d_ws;   // acc[0]=pos_score, acc[1]=neg_score
  float* out = (float*)d_out;

  colbert_init_kernel<<<1, 256, 0, stream>>>(acc);

  const int qtiles = B_DIM * (Q_DIM / 16);       // 4096
  const int blocks = qtiles / 4;                 // 4 q-tiles x {p,n} per block
  colbert_maxsim_kernel<<<blocks, 256, 0, stream>>>(q, p, n, acc);

  colbert_finalize_kernel<<<1, 1, 0, stream>>>(acc, out);
}
